// Net_LV_8658654069024
// MI455X (gfx1250) — matrix operand order repaired and checked by reading, not compiled, not run on hardware
//
#include <hip/hip_runtime.h>

typedef __attribute__((ext_vector_type(16))) _Float16 v16h;
typedef __attribute__((ext_vector_type(8)))  _Float16 v8h;
typedef __attribute__((ext_vector_type(2)))  _Float16 v2h;
typedef __attribute__((ext_vector_type(8)))  float    v8f;

namespace {
constexpr int   kMC      = 16384;
constexpr int   kSteps   = 120;
constexpr int   kPeriod  = 30;
constexpr int   kNmat    = 4;
constexpr int   kNK      = 21;
constexpr float kRate    = 0.05f;

constexpr int   kWaves   = 2;              // waves per block
constexpr int   kThreads = kWaves * 32;    // 64 threads / block
constexpr int   kPathsPB = kWaves * 16;    // 32 paths / block (16 per wave = WMMA M)
constexpr int   kBlocks  = kMC / kPathsPB; // 512 blocks
constexpr int   kPathPad = 128;            // f16 row stride: col0=t, 1..121=path, 122=ones(bias)

// workspace float offsets
constexpr int WS_PSUM = 0;    // 84: sum of price per (m,k)
constexpr int WS_PSQ  = 84;   // 84: sum of price^2
constexpr int WS_ESUM = 168;  // sum exotic_p
constexpr int WS_ESQ  = 169;  // sum exotic_p^2
constexpr int WS_DTE  = 170;  // sum discT*exotic
constexpr int WS_TOT  = 171;
}

// ---------------- WMMA fragment helpers (gfx1250 wave32 layouts, ISA 7.12.2) ----------------

// A fragment 16x32 f16 from row-major LDS: lane's 16 halves are row[k..k+7] ++ row[k+16..k+23]
__device__ __forceinline__ v16h load_fragA(const _Float16* A, int strideA, int kbase, int lane) {
  const int m  = lane & 15;
  const int hi = lane >> 4;
  const _Float16* row = A + m * strideA + kbase + hi * 8;
  v8h lo = *(const v8h*)(row);
  v8h hq = *(const v8h*)(row + 16);
  return __builtin_shufflevector(lo, hq, 0, 1, 2, 3, 4, 5, 6, 7, 8, 9, 10, 11, 12, 13, 14, 15);
}

// B fragment from LDS staged fragment-major: lane's 16 halves contiguous (32 B)
__device__ __forceinline__ v16h load_fragB(const _Float16* Wp, int nt, int c, int t, int lane) {
  const _Float16* p = Wp + (((c * nt + t) * 32 + lane) << 4);
  v8h lo = *(const v8h*)(p);
  v8h hq = *(const v8h*)(p + 8);
  return __builtin_shufflevector(lo, hq, 0, 1, 2, 3, 4, 5, 6, 7, 8, 9, 10, 11, 12, 13, 14, 15);
}

__device__ __forceinline__ v8f wmma_f16(v16h a, v16h b, v8f c) {
  return __builtin_amdgcn_wmma_f32_16x16x32_f16(false, a, false, b, (short)0, c, false, false);
}

// C fragment -> f16 convert, packed-f16 relu (v_cvt_pk + v_pk_max), scatter store
__device__ __forceinline__ void store_relu_f16(v8f acc, _Float16* O, int strideO, int n, int hi) {
  v8h hv;
#pragma unroll
  for (int r = 0; r < 8; ++r) hv[r] = (_Float16)acc[r];
  const v8h z8 = {};
  hv = __builtin_elementwise_max(hv, z8);
#pragma unroll
  for (int r = 0; r < 8; ++r) O[(r + 8 * hi) * strideO + n] = hv[r];  // row = r+8*hi, col = n
}

// One-wave GEMM layer with bias folded into the weights (ones-column): O = relu(A @ W)
template <int KPAD, int NPAD>
__device__ __forceinline__ void wave_gemm(const _Float16* A, int strideA,
                                          const _Float16* Wp,
                                          _Float16* O, int strideO, int lane) {
  constexpr int NT = NPAD / 16;
  constexpr int KC = KPAD / 32;
  const int hi = lane >> 4;
  v16h afr[KC];
#pragma unroll
  for (int c = 0; c < KC; ++c) afr[c] = load_fragA(A, strideA, c * 32, lane);
#pragma unroll
  for (int t = 0; t < NT; ++t) {
    v8f acc = {};
#pragma unroll
    for (int c = 0; c < KC; ++c)
      acc = wmma_f16(afr[c], load_fragB(Wp, NT, c, t, lane), acc);
    store_relu_f16(acc, O, strideO, (lane & 15) + 16 * t, hi);
  }
}

// ---------------- weight staging ----------------
// Global layout W[m][n][k]. Stage B-fragment-major f16 with:
//   row k==K   -> bias (multiplied by the ones-column of the input)
//   (K, ONESN) -> 1.0 so the ones feature propagates to the next layer
template <int K, int N, int KPAD, int NPAD, int ONESN>
__device__ __forceinline__ void stage_wp(_Float16* dst, const float* Wg, const float* bg,
                                         int m, int tid) {
  constexpr int NT = NPAD / 16;
  constexpr int total = KPAD * NPAD;
  for (int idx = tid; idx < total; idx += kThreads) {
    const int j  = idx & 15;
    const int L  = (idx >> 4) & 31;
    const int ft = idx >> 9;            // c*NT + t
    const int c  = ft / NT, t = ft - c * NT;
    const int k  = c * 32 + (j >> 3) * 16 + (L >> 4) * 8 + (j & 7);
    const int n  = (L & 15) + 16 * t;
    float v = 0.0f;
    if (k < K && n < N)            v = Wg[((size_t)m * N + n) * K + k];
    else if (k == K && n < N)      v = bg[(size_t)m * N + n];   // bias row
    else if (k == K && n == ONESN) v = 1.0f;                    // ones passthrough
    dst[idx] = (_Float16)v;
  }
}
__device__ __forceinline__ void stage_h(_Float16* dst, const float* src, int m, int count, int pad, int tid) {
  for (int i = tid; i < pad; i += kThreads)
    dst[i] = (_Float16)((i < count) ? src[(size_t)m * count + i] : 0.0f);
}
// de-interleave the K=2 layer weights into t-weights / S-weights
__device__ __forceinline__ void stage_pair(_Float16* dT, _Float16* dS, const float* Wg,
                                           int m, int N, int pad, int tid) {
  for (int n = tid; n < pad; n += kThreads) {
    dT[n] = (_Float16)((n < N) ? Wg[(size_t)m * N * 2 + n * 2]     : 0.0f);
    dS[n] = (_Float16)((n < N) ? Wg[(size_t)m * N * 2 + n * 2 + 1] : 0.0f);
  }
}

// split-dot: half-wave g-interleaved, combine with shfl_xor(16); bias folded at ones index
__device__ __forceinline__ float half_dot(const _Float16* row, const _Float16* w, int groups, int hi) {
  float u = 0.0f;
  for (int g = hi; g < groups; g += 2) {
    v8h x  = *(const v8h*)(row + 8 * g);
    v8h ww = *(const v8h*)(w + 8 * g);
#if __has_builtin(__builtin_amdgcn_fdot2)
#pragma unroll
    for (int q = 0; q < 4; ++q) {
      v2h xp = __builtin_shufflevector(x,  x,  2 * q, 2 * q + 1);
      v2h wp = __builtin_shufflevector(ww, ww, 2 * q, 2 * q + 1);
      u = __builtin_amdgcn_fdot2(xp, wp, u, false);     // v_dot2_f32_f16
    }
#else
#pragma unroll
    for (int j = 0; j < 8; ++j) u += (float)x[j] * (float)ww[j];
#endif
  }
  return u;
}

struct LVParams {
  const float *S0, *z;
  const float *dpW1, *dpB1, *dpW2, *dpB2, *dpW3, *dpB3;
  const float *vpW1, *vpB1, *vpW2, *vpB2, *vpW3, *vpB3, *vpW4, *vpB4;
  const float *epW1, *epB1, *epW2, *epB2, *epW3, *epB3, *epW4, *epB4;
  float *out_path, *out_diff, *out_exp;
  float *ws;
};

__global__ void lv_ws_init(float* ws) {
  int g = blockIdx.x * blockDim.x + threadIdx.x;
  if (g < WS_TOT) ws[g] = 0.0f;
}

__global__ __launch_bounds__(kThreads)
void lv_sim(LVParams P) {
  // ---- per-segment weights (B-fragment-major f16, bias folded) ----
  __shared__ _Float16 w_dp2[64 * 64];
  __shared__ _Float16 w_vp2[32 * 32];
  __shared__ _Float16 w_vp3[32 * 32];
  __shared__ _Float16 w_vp4[32 * 96];
  __shared__ _Float16 w_ep1[128 * 32];
  __shared__ _Float16 w_ep2[32 * 32];
  __shared__ _Float16 w_ep3[32 * 32];
  // K=2 input layers (packed f16, de-interleaved) and 1-wide output dots
  __shared__ _Float16 w_dp1t[64], w_dp1S[64], b_dp1h[64], w_dp3h[64];
  __shared__ _Float16 w_vp1t[32], w_vp1S[32], b_vp1h[32], w_ep4h[32];
  // running path as f16 ep-input tile: col0=t, 1..121=path, 122=ones
  __shared__ _Float16 pathh[kPathsPB * kPathPad];
  // per-wave activation ping-pong
  __shared__ _Float16 actA[kWaves][16 * 64];
  __shared__ _Float16 actB[kWaves][16 * 64];
  // per-path state
  __shared__ float Sshr[kPathsPB], rmaxshr[kPathsPB], diffshr[kPathsPB];
  __shared__ float coefS[kPathsPB], dWshr[kPathsPB], cvexoshr[kPathsPB];
  // reductions
  __shared__ float ps[kNK], pq[kNK], red[3];

  const int tid  = threadIdx.x;
  const int lane = tid & 31;
  const int wv   = tid >> 5;
  const int m16  = lane & 15;
  const int hi   = lane >> 4;
  const int pbase = wv * 16;
  const int gbase = blockIdx.x * kPathsPB;

  const float hh  = 1.0f / 120.0f;
  const float sqh = sqrtf(hh);
  const float S0v = P.S0[0];

  // ---- init ----
  for (int idx = tid; idx < kPathsPB * kPathPad; idx += kThreads) pathh[idx] = (_Float16)0.0f;
  if (tid < 3) red[tid] = 0.0f;
  if (hi == 0) {
    const int p = pbase + m16;
    Sshr[p] = S0v; rmaxshr[p] = S0v; diffshr[p] = 0.0f; cvexoshr[p] = 0.0f;
    P.out_path[(size_t)(gbase + p) * (kSteps + 1)] = S0v;
  }
  __syncthreads();
  if (hi == 0) {
    pathh[(pbase + m16) * kPathPad + 1]   = (_Float16)S0v;   // path[0] = S0
    pathh[(pbase + m16) * kPathPad + 122] = (_Float16)1.0f;  // ones feature (ep bias row)
  }

  // cv_van lives permanently in the vp-output accumulator distribution
  v8f cvv_acc[6];
  {
    v8f zz = {};
#pragma unroll
    for (int t = 0; t < 6; ++t) cvv_acc[t] = zz;
  }

  for (int i = 0; i < kSteps; ++i) {
    const int mseg = i / kPeriod;
    if ((i % kPeriod) == 0) {   // stage this segment's weights (cross-wave -> barriers)
      __syncthreads();
      stage_wp<50, 50, 64, 64, 50>(w_dp2, P.dpW2, P.dpB2, mseg, tid);
      stage_wp<30, 30, 32, 32, 30>(w_vp2, P.vpW2, P.vpB2, mseg, tid);
      stage_wp<30, 30, 32, 32, 30>(w_vp3, P.vpW3, P.vpB3, mseg, tid);
      stage_wp<30, 84, 32, 96, -1>(w_vp4, P.vpW4, P.vpB4, mseg, tid);
      stage_wp<122, 20, 128, 32, 20>(w_ep1, P.epW1, P.epB1, mseg, tid);
      stage_wp<20, 20, 32, 32, 20>(w_ep2, P.epW2, P.epB2, mseg, tid);
      stage_wp<20, 20, 32, 32, 20>(w_ep3, P.epW3, P.epB3, mseg, tid);
      stage_pair(w_dp1t, w_dp1S, P.dpW1, mseg, 50, 64, tid);
      stage_h(b_dp1h, P.dpB1, mseg, 50, 64, tid);
      stage_h(w_dp3h, P.dpW3, mseg, 50, 64, tid);
      stage_pair(w_vp1t, w_vp1S, P.vpW1, mseg, 30, 32, tid);
      stage_h(b_vp1h, P.vpB1, mseg, 30, 32, tid);
      stage_h(w_ep4h, P.epW4, mseg, 20, 32, tid);
      if (tid == 0) {
        b_dp1h[50] = (_Float16)1.0f;          // dp ones feature
        b_vp1h[30] = (_Float16)1.0f;          // vp ones feature
        w_dp3h[50] = (_Float16)P.dpB3[mseg];  // dp output bias at ones index
        w_ep4h[20] = (_Float16)P.epB4[mseg];  // ep output bias at ones index
      }
      __syncthreads();
    }

    const float tp   = (float)i * hh;            // timegrid[i]
    const float disc = __expf(-kRate * tp);
    if (hi == 0) {
      pathh[(pbase + m16) * kPathPad + 0] = (_Float16)tp;    // t feature for ep net
      if (i + 1 < kSteps)                                     // gfx1250 global_prefetch
        __builtin_prefetch(&P.z[(size_t)(gbase + pbase + m16) * kSteps + i + 1], 0, 0);
    }
    const float Sp0 = Sshr[pbase + m16];
    const _Float16 tph = (_Float16)tp;
    const _Float16 Sph = (_Float16)Sp0;
    const v8h z8 = {};

    // ---- dp layer1 (K=2): [t,S] -> 50 relu + ones col, packed f16 math ----
    {
      _Float16* orow = &actA[wv][m16 * 64];
#pragma unroll
      for (int g = hi; g < 8; g += 2) {
        v8h wt = *(const v8h*)(w_dp1t + 8 * g);
        v8h wS = *(const v8h*)(w_dp1S + 8 * g);
        v8h bb = *(const v8h*)(b_dp1h + 8 * g);
        v8h r  = wt * tph + wS * Sph + bb;             // v_pk_fma_f16
        *(v8h*)(orow + 8 * g) = __builtin_elementwise_max(r, z8);
      }
    }
    // dp layer2: 50(+1)->50(+1) (8 WMMA, bias folded)
    wave_gemm<64, 64>(&actA[wv][0], 64, w_dp2, &actB[wv][0], 64, lane);
    // dp layer3: split-dot (bias at ones index) + softplus ; CV coefficient
    {
      const float part = half_dot(&actB[wv][m16 * 64], w_dp3h, 8, hi);
      const float u = part + __shfl_xor(part, 16);
      if (hi == 0) {
        const int p = pbase + m16;
        const float df = (u > 20.0f) ? u : log1pf(__expf(u));     // softplus
        diffshr[p] = df;
        const float zc = P.z[(size_t)(gbase + p) * kSteps + i];
        const float dW = sqh * zc;
        dWshr[p] = dW;
        coefS[p] = disc * Sp0 * df * dW;
      }
    }

    // ---- vp layer1 (K=2): [t,S] -> 30 relu + ones col ----
    {
      _Float16* orow = &actA[wv][m16 * 64];
#pragma unroll
      for (int g = hi; g < 4; g += 2) {
        v8h wt = *(const v8h*)(w_vp1t + 8 * g);
        v8h wS = *(const v8h*)(w_vp1S + 8 * g);
        v8h bb = *(const v8h*)(b_vp1h + 8 * g);
        v8h r  = wt * tph + wS * Sph + bb;
        *(v8h*)(orow + 8 * g) = __builtin_elementwise_max(r, z8);
      }
    }
    wave_gemm<32, 32>(&actA[wv][0], 64, w_vp2, &actB[wv][0], 64, lane);   // 2 WMMA
    wave_gemm<32, 32>(&actB[wv][0], 64, w_vp3, &actA[wv][0], 64, lane);   // 2 WMMA
    // vp layer4: 30->84 (bias folded); cv_van += coef * acc  (6 WMMA)
    {
      float c8[8];
#pragma unroll
      for (int r = 0; r < 8; ++r) c8[r] = coefS[pbase + r + 8 * hi];
      v16h a = load_fragA(&actA[wv][0], 64, 0, lane);
#pragma unroll
      for (int t = 0; t < 6; ++t) {
        v8f acc = {};
        acc = wmma_f16(a, load_fragB(w_vp4, 6, 0, t, lane), acc);
#pragma unroll
        for (int r = 0; r < 8; ++r) cvv_acc[t][r] += c8[r] * acc[r];
      }
    }

    // ---- ep layer1: [t, path, 1] K=123 pad 128, dynamic-K: skip chunks of still-zero path ----
    {
      const _Float16* A = &pathh[pbase * kPathPad];
      const int cmax = (i + 2) >> 5;               // highest chunk holding nonzero path data
      v16h a0 = load_fragA(A, kPathPad, 0, lane);
      v16h a3 = load_fragA(A, kPathPad, 96, lane); // always: holds ones col 122 (bias row)
      v16h a1 = {}, a2 = {};
      if (cmax >= 1) a1 = load_fragA(A, kPathPad, 32, lane);
      if (cmax >= 2) a2 = load_fragA(A, kPathPad, 64, lane);
#pragma unroll
      for (int t = 0; t < 2; ++t) {
        v8f acc = {};
        acc = wmma_f16(a0, load_fragB(w_ep1, 2, 0, t, lane), acc);
        if (cmax >= 1) acc = wmma_f16(a1, load_fragB(w_ep1, 2, 1, t, lane), acc);
        if (cmax >= 2) acc = wmma_f16(a2, load_fragB(w_ep1, 2, 2, t, lane), acc);
        acc = wmma_f16(a3, load_fragB(w_ep1, 2, 3, t, lane), acc);
        store_relu_f16(acc, &actB[wv][0], 64, (lane & 15) + 16 * t, hi);
      }
    }
    wave_gemm<32, 32>(&actB[wv][0], 64, w_ep2, &actA[wv][0], 64, lane);   // 2 WMMA
    wave_gemm<32, 32>(&actA[wv][0], 64, w_ep3, &actB[wv][0], 64, lane);   // 2 WMMA
    // ep layer4 split-dot, cv_exo update, S update, path write
    {
      const float part = half_dot(&actB[wv][m16 * 64], w_ep4h, 4, hi);
      const float cve = part + __shfl_xor(part, 16);
      if (hi == 0) {
        const int p = pbase + m16;
        cvexoshr[p] += coefS[p] * cve;
        const float df = diffshr[p], dW = dWshr[p];
        const float Snew = Sp0 + kRate * Sp0 * hh / (1.0f + kRate * Sp0 * sqh)
                         + Sp0 * df * dW / (1.0f + Sp0 * df * sqh);
        Sshr[p] = Snew;
        rmaxshr[p] = fmaxf(rmaxshr[p], Snew);
        pathh[p * kPathPad + (i + 2)] = (_Float16)Snew;                  // path[i+1]
        P.out_path[(size_t)(gbase + p) * (kSteps + 1) + (i + 1)] = Snew;
      }
    }

    // ---- segment end: vanilla price statistics for maturity mseg (cross-wave) ----
    if ((i % kPeriod) == (kPeriod - 1)) {
      for (int k2 = tid; k2 < 2 * kNK; k2 += kThreads) {
        if (k2 < kNK) ps[k2] = 0.0f; else pq[k2 - kNK] = 0.0f;
      }
      __syncthreads();
      const float discm = __expf(-kRate * (float)(kPeriod * (mseg + 1)) * hh);
#pragma unroll
      for (int t = 0; t < 6; ++t) {
        const int c = (lane & 15) + 16 * t;        // global cv_van column
        const int k = c - mseg * kNK;
        if (k >= 0 && k < kNK) {
          const float strike = 0.8f + 0.02f * (float)k;
          float lsum = 0.0f, lsq = 0.0f;
#pragma unroll
          for (int r = 0; r < 8; ++r) {
            const float Sp = Sshr[pbase + r + 8 * hi];
            const float pr = discm * fmaxf(Sp - strike, 0.0f) - cvv_acc[t][r];
            lsum += pr; lsq += pr * pr;
          }
          atomicAdd(&ps[k], lsum);
          atomicAdd(&pq[k], lsq);
        }
      }
      __syncthreads();
      for (int k2 = tid; k2 < kNK; k2 += kThreads) {
        atomicAdd(&P.ws[WS_PSUM + mseg * kNK + k2], ps[k2]);
        atomicAdd(&P.ws[WS_PSQ  + mseg * kNK + k2], pq[k2]);
      }
      __syncthreads();
    }
  }

  // ---- epilogue: exotic payoff, diffusion, reductions ----
  const float discT = __expf(-kRate * 1.0f);
  if (hi == 0) {
    const int p = pbase + m16;
    const float exo = rmaxshr[p] - Sshr[p];
    const float dte = discT * exo;
    const float epay = dte - cvexoshr[p];        // exotic_p
    P.out_diff[gbase + p] = diffshr[p];
    P.out_exp[gbase + p]  = epay;
    atomicAdd(&red[0], epay);
    atomicAdd(&red[1], epay * epay);
    atomicAdd(&red[2], dte);
  }
  __syncthreads();
  if (tid == 0) {
    atomicAdd(&P.ws[WS_ESUM], red[0]);
    atomicAdd(&P.ws[WS_ESQ],  red[1]);
    atomicAdd(&P.ws[WS_DTE],  red[2]);
  }
}

__global__ void lv_finalize(const float* ws, const float* exp_p,
                            float* out_pcv, float* out_vcv,
                            float* out_mean, float* out_var, float* out_err) {
  const int g = blockIdx.x * blockDim.x + threadIdx.x;
  const float invMC = 1.0f / (float)kMC;
  const float invM1 = 1.0f / (float)(kMC - 1);
  if (g < kNmat * kNK) {
    const float s = ws[WS_PSUM + g], q = ws[WS_PSQ + g];
    out_pcv[g] = s * invMC;
    out_vcv[g] = (q - s * s * invMC) * invM1;
  }
  if (g == 0) {
    const float s = ws[WS_ESUM], q = ws[WS_ESQ];
    out_mean[0] = s * invMC;
    out_var[0]  = (q - s * s * invMC) * invM1;
  }
  if (g < kMC) {
    out_err[g] = exp_p[g] - ws[WS_DTE] * invMC;   // error = exotic_p - mean(discT*exotic)
  }
}

extern "C" void kernel_launch(void* const* d_in, const int* in_sizes, int n_in,
                              void* d_out, int out_size, void* d_ws, size_t ws_size,
                              hipStream_t stream) {
  (void)in_sizes; (void)n_in; (void)out_size; (void)ws_size;
  // setup_inputs() pytree leaves, dict order, depth-first:
  // 0:S0 1:z 2:timegrid 3..8:dp(W1,b1,W2,b2,W3,b3) 9..16:vp(W1..b4) 17..24:ep(W1..b4) 25:MC 26:ind_T
  LVParams P;
  P.S0   = (const float*)d_in[0];
  P.z    = (const float*)d_in[1];
  P.dpW1 = (const float*)d_in[3];  P.dpB1 = (const float*)d_in[4];
  P.dpW2 = (const float*)d_in[5];  P.dpB2 = (const float*)d_in[6];
  P.dpW3 = (const float*)d_in[7];  P.dpB3 = (const float*)d_in[8];
  P.vpW1 = (const float*)d_in[9];  P.vpB1 = (const float*)d_in[10];
  P.vpW2 = (const float*)d_in[11]; P.vpB2 = (const float*)d_in[12];
  P.vpW3 = (const float*)d_in[13]; P.vpB3 = (const float*)d_in[14];
  P.vpW4 = (const float*)d_in[15]; P.vpB4 = (const float*)d_in[16];
  P.epW1 = (const float*)d_in[17]; P.epB1 = (const float*)d_in[18];
  P.epW2 = (const float*)d_in[19]; P.epB2 = (const float*)d_in[20];
  P.epW3 = (const float*)d_in[21]; P.epB3 = (const float*)d_in[22];
  P.epW4 = (const float*)d_in[23]; P.epB4 = (const float*)d_in[24];

  float* o = (float*)d_out;
  float* out_path = o;                                  // 16384*121
  float* out_diff = out_path + (size_t)kMC * (kSteps + 1);
  float* out_pcv  = out_diff + kMC;                     // 84
  float* out_vcv  = out_pcv + kNmat * kNK;              // 84
  float* out_exp  = out_vcv + kNmat * kNK;              // 16384
  float* out_mean = out_exp + kMC;                      // 1
  float* out_var  = out_mean + 1;                       // 1
  float* out_err  = out_var + 1;                        // 16384

  P.out_path = out_path;
  P.out_diff = out_diff;
  P.out_exp  = out_exp;
  P.ws       = (float*)d_ws;

  hipLaunchKernelGGL(lv_ws_init, dim3(1), dim3(256), 0, stream, P.ws);
  hipLaunchKernelGGL(lv_sim, dim3(kBlocks), dim3(kThreads), 0, stream, P);
  hipLaunchKernelGGL(lv_finalize, dim3((kMC + 255) / 256), dim3(256), 0, stream,
                     P.ws, out_exp, out_pcv, out_vcv, out_mean, out_var, out_err);
}
